// SelfAttention_1460288880733
// MI455X (gfx1250) — matrix operand order repaired and checked by reading, not compiled, not run on hardware
//
#include <hip/hip_runtime.h>

// ---------------------------------------------------------------------------
// Self-attention (B=4, N=4096, C=256, CK=32) for MI455X (gfx1250, wave32).
// Flash-attention structure: g/hh stream from L2 with double-buffered
// (async-to-LDS when available) staging, online softmax, all GEMMs on
// v_wmma_f32_16x16x32_bf16 with fp32 accumulation.
// ---------------------------------------------------------------------------

typedef __attribute__((ext_vector_type(16))) __bf16 v16bf;
typedef __attribute__((ext_vector_type(8)))  __bf16 v8bf;
typedef __attribute__((ext_vector_type(8)))  float  v8f;
typedef __attribute__((ext_vector_type(4)))  int    v4i;

#define NPIX 4096          // H*W
#define CDIM 256           // channels
#define ROWFG 64           // fused f|g row (bf16)
#define HHP  40            // padded key stride of hh in LDS (keeps b128 align,
                           // reduces bank conflicts)

#if defined(__has_builtin)
#  if __has_builtin(__builtin_amdgcn_global_load_async_to_lds_b128)
#    define HAVE_ASYNC_LDS 1
#  endif
#endif
#ifndef HAVE_ASYNC_LDS
#  define HAVE_ASYNC_LDS 0
#endif

__device__ __forceinline__ unsigned short f32_to_bf16(float f) {
  union { float f; unsigned int u; } v; v.f = f;
  unsigned int r = v.u + 0x7FFFu + ((v.u >> 16) & 1u);   // RNE
  return (unsigned short)(r >> 16);
}

union ABu { v16bf v; v8bf h[2]; };

// Build a 16-element bf16 operand from two contiguous 16B chunks.
__device__ __forceinline__ v16bf load_ab(const unsigned short* p0,
                                         const unsigned short* p1) {
  ABu t;
  t.h[0] = *(const v8bf*)p0;
  t.h[1] = *(const v8bf*)p1;
  return t.v;
}

__device__ __forceinline__ v8f wmma_bf16(v16bf a, v16bf b, v8f c) {
  return __builtin_amdgcn_wmma_f32_16x16x32_bf16(
      /*neg_a=*/false, a, /*neg_b=*/false, b,
      /*c_mod=*/(short)0, c, /*reuse_a=*/false, /*reuse_b=*/false);
}

// 16-byte global -> LDS copy; async (ASYNCcnt-tracked DMA) when available.
__device__ __forceinline__ void cp128(const unsigned short* g,
                                      unsigned short* l) {
#if HAVE_ASYNC_LDS
  __builtin_amdgcn_global_load_async_to_lds_b128(
      (__attribute__((address_space(1))) v4i*)(g),
      (__attribute__((address_space(3))) v4i*)(l), 0, 0);
#else
  *(uint4*)l = *(const uint4*)g;
#endif
}

// ---------------------------------------------------------------------------
// Kernel 1: fp32 -> bf16 conversion of x; build WbT[320][256] = [Wf|Wg|Wh]^T
// (column-major so WMMA B-operands read 16 contiguous K values per lane);
// concatenate biases.
// ---------------------------------------------------------------------------
__global__ __launch_bounds__(256) void prep_kernel(
    const float* __restrict__ x,
    const float* __restrict__ Wf, const float* __restrict__ bfv,
    const float* __restrict__ Wg, const float* __restrict__ bgv,
    const float* __restrict__ Wh, const float* __restrict__ bhv,
    unsigned short* __restrict__ xb, unsigned short* __restrict__ WbT,
    float* __restrict__ bcat) {
  int i0 = blockIdx.x * blockDim.x + threadIdx.x;
  int gs = gridDim.x * blockDim.x;
  for (int i = i0; i < 4 * NPIX * CDIM; i += gs) xb[i] = f32_to_bf16(x[i]);
  for (int i = i0; i < 320 * CDIM; i += gs) {
    int j = i / CDIM, k = i % CDIM;   // j = fused output col, k = input channel
    float v = (j < 32) ? Wf[k * 32 + j]
            : (j < 64) ? Wg[k * 32 + (j - 32)]
                       : Wh[k * CDIM + (j - 64)];
    WbT[j * CDIM + k] = f32_to_bf16(v);
  }
  for (int i = i0; i < 320; i += gs)
    bcat[i] = (i < 32) ? bfv[i] : (i < 64) ? bgv[i - 32] : bhv[i - 64];
}

// ---------------------------------------------------------------------------
// Kernel 2: fused projection GEMM: [16384,256] @ [256,320] + b.
// Columns 0-63 (f|g) -> pfg[16384,64] row-major.
// Columns 64-319 (hh) -> hhT[4,256,4096] channel-major (so the attention
// kernel can stage its 32-key chunks with coalesced b128 loads that land in
// the contiguous-K layout this code uses for its WMMA B operands).
// ---------------------------------------------------------------------------
__global__ __launch_bounds__(256) void proj_kernel(
    const unsigned short* __restrict__ xb,
    const unsigned short* __restrict__ WbT,
    const float* __restrict__ bcat,
    unsigned short* __restrict__ pfg,
    unsigned short* __restrict__ hhT) {
  int tid = threadIdx.x;
  int w = tid >> 5, lane = tid & 31, h = lane >> 4, ln = lane & 15;
  int gw = blockIdx.x * 8 + w;          // 20480 tiles total
  int mt = gw / 20, nt = gw % 20;
  int row0 = mt * 16, n0 = nt * 16;

  v8f acc = {};
  const unsigned short* arow = xb + (size_t)(row0 + ln) * CDIM;
  const unsigned short* bcol = WbT + (size_t)(n0 + ln) * CDIM;
#pragma unroll
  for (int kk = 0; kk < CDIM; kk += 32) {
    // A lane: row = ln; elems = 16 contiguous K at kk+h*16
    v16bf a = load_ab(arow + kk + h * 16, arow + kk + h * 16 + 8);
    // B lane: col = ln; elems = 16 contiguous K at kk+h*16
    v16bf b = load_ab(bcol + kk + h * 16, bcol + kk + h * 16 + 8);
    acc = wmma_bf16(a, b, acc);
  }
  float bv = bcat[n0 + ln];
  if (nt < 4) {                          // f|g block
#pragma unroll
    for (int r = 0; r < 8; r++) {
      int row = row0 + 8 * h + r;        // C layout: M = r + 8*(lane>=16)
      pfg[(size_t)row * ROWFG + n0 + ln] = f32_to_bf16(acc[r] + bv);
    }
  } else {                               // hh block, store transposed
    int ch = n0 - 64 + ln;
#pragma unroll
    for (int r = 0; r < 8; r++) {
      int row = row0 + 8 * h + r;        // row = b*4096 + q
      int bb = row >> 12, q = row & 4095;
      hhT[((size_t)bb * CDIM + ch) * NPIX + q] = f32_to_bf16(acc[r] + bv);
    }
  }
}

// ---------------------------------------------------------------------------
// Kernel 3: flash attention. Block = 32 queries x 256 channels, 8 waves:
// wave = (query tile qt in {0,1}, channel group cg in {0..3} of 64 channels).
// Keys streamed in chunks of 32 with double-buffered async LDS staging.
// ---------------------------------------------------------------------------
__global__ __launch_bounds__(256) void attn_kernel(
    const unsigned short* __restrict__ pfg,
    const unsigned short* __restrict__ hhT,
    const float* __restrict__ x,
    const float* __restrict__ gamma,
    float* __restrict__ out) {
  __shared__ __align__(16) unsigned short g_lds[2][32 * 32];      // [key][feat]
  __shared__ __align__(16) unsigned short hh_lds[2][CDIM * HHP];  // [ch][key]
  __shared__ __align__(16) unsigned short P_lds[8 * 16 * 32];     // per-wave P

  int tid = threadIdx.x, w = tid >> 5, lane = tid & 31;
  int h = lane >> 4, ln = lane & 15;
  int b = blockIdx.x >> 7, qblk = blockIdx.x & 127;
  int qt = w >> 2, cg = w & 3;
  int q0 = qblk * 32 + qt * 16;
  int chb = cg * 64;
  const size_t bbase = (size_t)b * NPIX;
  const unsigned short* hhb = hhT + (size_t)b * CDIM * NPIX;

  // Stage one 32-key chunk (g: 2 KB, hh: 16 KB) into LDS buffer nb.
  auto stage = [&](int kk, int nb) {
    if (tid < 128) {                      // g: 128 x b128
      int k = tid >> 2, c8 = (tid & 3) * 8;
      cp128(pfg + (bbase + kk + k) * ROWFG + 32 + c8, &g_lds[nb][k * 32 + c8]);
    }
#pragma unroll
    for (int i = 0; i < 4; i++) {         // hh: 1024 x b128, coalesced
      int j = tid + 256 * i;
      int ch = j >> 2, k8 = (j & 3) * 8;
      cp128(hhb + (size_t)ch * NPIX + kk + k8, &hh_lds[nb][ch * HHP + k8]);
    }
  };

  // Persistent A operand: f rows for this wave's 16 queries.
  const unsigned short* frow = pfg + (bbase + q0 + ln) * ROWFG;
  v16bf fa = load_ab(frow + h * 16, frow + h * 16 + 8);

  v8f acc0 = {}, acc1 = {}, acc2 = {}, acc3 = {};
  float m[8], l[8];
#pragma unroll
  for (int r = 0; r < 8; r++) { m[r] = -3.0e38f; l[r] = 0.0f; }

  unsigned short* Pw = P_lds + w * 512;

  stage(0, 0);                            // prologue

  for (int it = 0; it < NPIX / 32; it++) {
    int cb = it & 1;
    // Our async copies from the previous iteration must land before the
    // barrier lets any wave read this buffer.
    asm volatile("s_wait_asynccnt 0" ::: "memory");
    __syncthreads();
    if (it + 1 < NPIX / 32) stage((it + 1) * 32, cb ^ 1);  // overlap with compute

    // s = f @ g^T : two 16x16 tiles (keys 0-15, 16-31 of this chunk).
    const unsigned short* gb = &g_lds[cb][0];
    v16bf bg0 = load_ab(gb + ln * 32 + h * 16, gb + ln * 32 + h * 16 + 8);
    v16bf bg1 = load_ab(gb + (16 + ln) * 32 + h * 16,
                        gb + (16 + ln) * 32 + h * 16 + 8);
    v8f z = {};
    v8f s0 = wmma_bf16(fa, bg0, z);
    v8f s1 = wmma_bf16(fa, bg1, z);

    // Online softmax. Row r of C tile = query 8*h + r; columns across 16 lanes.
    float rmax[8], mnew[8], alpha[8], p0v[8], p1v[8], rsum[8];
#pragma unroll
    for (int r = 0; r < 8; r++) rmax[r] = fmaxf(s0[r], s1[r]);
#pragma unroll
    for (int off = 1; off < 16; off <<= 1) {
#pragma unroll
      for (int r = 0; r < 8; r++)
        rmax[r] = fmaxf(rmax[r], __shfl_xor(rmax[r], off, 16));
    }
#pragma unroll
    for (int r = 0; r < 8; r++) {
      mnew[r] = fmaxf(m[r], rmax[r]);
      alpha[r] = __expf(m[r] - mnew[r]);
      p0v[r] = __expf(s0[r] - mnew[r]);
      p1v[r] = __expf(s1[r] - mnew[r]);
      rsum[r] = p0v[r] + p1v[r];
    }
#pragma unroll
    for (int off = 1; off < 16; off <<= 1) {
#pragma unroll
      for (int r = 0; r < 8; r++) rsum[r] += __shfl_xor(rsum[r], off, 16);
    }
#pragma unroll
    for (int r = 0; r < 8; r++) {
      l[r] = l[r] * alpha[r] + rsum[r];
      m[r] = mnew[r];
      acc0[r] *= alpha[r]; acc1[r] *= alpha[r];
      acc2[r] *= alpha[r]; acc3[r] *= alpha[r];
    }

    // P (bf16) -> per-wave LDS tile, row-major [16 q][32 k].
#pragma unroll
    for (int r = 0; r < 8; r++) {
      Pw[(8 * h + r) * 32 + ln] = f32_to_bf16(p0v[r]);
      Pw[(8 * h + r) * 32 + 16 + ln] = f32_to_bf16(p1v[r]);
    }
    asm volatile("s_wait_dscnt 0" ::: "memory");  // intra-wave LDS RAW

    // Re-read P in A-operand layout; accumulate O += P @ hh over 4 ch tiles.
    v16bf pa = load_ab(Pw + ln * 32 + h * 16, Pw + ln * 32 + h * 16 + 8);
    const unsigned short* hb = &hh_lds[cb][0];
    v16bf b0 = load_ab(hb + (chb + ln) * HHP + h * 16,
                       hb + (chb + ln) * HHP + h * 16 + 8);
    acc0 = wmma_bf16(pa, b0, acc0);
    v16bf b1 = load_ab(hb + (chb + 16 + ln) * HHP + h * 16,
                       hb + (chb + 16 + ln) * HHP + h * 16 + 8);
    acc1 = wmma_bf16(pa, b1, acc1);
    v16bf b2 = load_ab(hb + (chb + 32 + ln) * HHP + h * 16,
                       hb + (chb + 32 + ln) * HHP + h * 16 + 8);
    acc2 = wmma_bf16(pa, b2, acc2);
    v16bf b3 = load_ab(hb + (chb + 48 + ln) * HHP + h * 16,
                       hb + (chb + 48 + ln) * HHP + h * 16 + 8);
    acc3 = wmma_bf16(pa, b3, acc3);
  }

  // Finalize: out = gamma * (O / l) + x
  float g0 = gamma[0];
  float inv[8];
#pragma unroll
  for (int r = 0; r < 8; r++) inv[r] = 1.0f / l[r];

#define FINAL_TILE(ACC, T)                                              \
  _Pragma("unroll")                                                     \
  for (int r = 0; r < 8; r++) {                                         \
    size_t n = bbase + (size_t)(q0 + 8 * h + r);                        \
    size_t idx = n * CDIM + (size_t)(chb + (T) * 16 + ln);              \
    out[idx] = g0 * (ACC[r] * inv[r]) + x[idx];                         \
  }
  FINAL_TILE(acc0, 0)
  FINAL_TILE(acc1, 1)
  FINAL_TILE(acc2, 2)
  FINAL_TILE(acc3, 3)
#undef FINAL_TILE
}

// ---------------------------------------------------------------------------
// Launch
// ---------------------------------------------------------------------------
extern "C" void kernel_launch(void* const* d_in, const int* in_sizes, int n_in,
                              void* d_out, int out_size, void* d_ws, size_t ws_size,
                              hipStream_t stream) {
  (void)in_sizes; (void)n_in; (void)out_size; (void)ws_size;
  const float* x    = (const float*)d_in[0];
  const float* Wf   = (const float*)d_in[1];
  const float* bfv  = (const float*)d_in[2];
  const float* Wg   = (const float*)d_in[3];
  const float* bgv  = (const float*)d_in[4];
  const float* Wh   = (const float*)d_in[5];
  const float* bhv  = (const float*)d_in[6];
  const float* gam  = (const float*)d_in[7];

  char* ws = (char*)d_ws;
  unsigned short* xb   = (unsigned short*)(ws);              //  8,388,608 B
  unsigned short* WbT  = (unsigned short*)(ws + 8388608);    //    163,840 B
  float*          bcat = (float*)(ws + 8552448);             //      1,280 B
  unsigned short* pfg  = (unsigned short*)(ws + 8553728);    //  2,097,152 B
  unsigned short* hhT  = (unsigned short*)(ws + 10650880);   //  8,388,608 B

  float* out = (float*)d_out;

  prep_kernel<<<1024, 256, 0, stream>>>(x, Wf, bfv, Wg, bgv, Wh, bhv,
                                        xb, WbT, bcat);
  proj_kernel<<<2560, 256, 0, stream>>>(xb, WbT, bcat, pfg, hhT);
  attn_kernel<<<512, 256, 0, stream>>>(pfg, hhT, x, gam, out);
}
